// NodeModel_5695126634531
// MI455X (gfx1250) — matrix operand order repaired and checked by reading, not compiled, not run on hardware
//
#include <hip/hip_runtime.h>

typedef __attribute__((ext_vector_type(16))) __bf16 v16bf;
typedef __attribute__((ext_vector_type(8)))  float  v8f;

#define F_NODE 32
#define F_EDGE 32
#define HID    64
#define N_TGT  32

// Native f32 -> bf16 (hardware convert on gfx1250, RNE)
__device__ __forceinline__ unsigned short f2bf(float f) {
    return __builtin_bit_cast(unsigned short, (__bf16)f);
}

struct B32x2 { uint4 a, b; };

// 16 contiguous bf16 from LDS (32B, 16B-aligned) -> B fragment
__device__ __forceinline__ v16bf ld16(const unsigned short* p) {
    B32x2 t; t.a = *(const uint4*)p; t.b = *(const uint4*)(p + 8);
    return __builtin_bit_cast(v16bf, t);
}
// two separate 8-bf16 chunks -> A fragment (K groups 0..7 / 16..23 pattern)
__device__ __forceinline__ v16bf ld8_8(const unsigned short* p0, const unsigned short* p1) {
    B32x2 t; t.a = *(const uint4*)p0; t.b = *(const uint4*)p1;
    return __builtin_bit_cast(v16bf, t);
}

// Build A fragment (16x32 bf16) for this lane from a 32-float row in global
// memory. Lane covers M = lane&15; elements 0..7 -> K = half*8+0..7,
// elements 8..15 -> K = 16+half*8+0..7  (half = lane>>4).
__device__ __forceinline__ v16bf gather_row_bf16(const float* row, int half, float scale) {
    const float* p = row + half * 8;
    float4 q0 = *(const float4*)(p);
    float4 q1 = *(const float4*)(p + 4);
    float4 q2 = *(const float4*)(p + 16);
    float4 q3 = *(const float4*)(p + 20);
    v16bf u;
    u[0]  = (__bf16)(q0.x * scale); u[1]  = (__bf16)(q0.y * scale);
    u[2]  = (__bf16)(q0.z * scale); u[3]  = (__bf16)(q0.w * scale);
    u[4]  = (__bf16)(q1.x * scale); u[5]  = (__bf16)(q1.y * scale);
    u[6]  = (__bf16)(q1.z * scale); u[7]  = (__bf16)(q1.w * scale);
    u[8]  = (__bf16)(q2.x * scale); u[9]  = (__bf16)(q2.y * scale);
    u[10] = (__bf16)(q2.z * scale); u[11] = (__bf16)(q2.w * scale);
    u[12] = (__bf16)(q3.x * scale); u[13] = (__bf16)(q3.y * scale);
    u[14] = (__bf16)(q3.z * scale); u[15] = (__bf16)(q3.w * scale);
    return u;
}

__device__ __forceinline__ v8f wmma_bf16(v16bf a, v16bf b, v8f c) {
    return __builtin_amdgcn_wmma_f32_16x16x32_bf16(false, a, false, b, (short)0, c, false, false);
}

__global__ void zero_ws(float* p, size_t n) {
    size_t i = (size_t)blockIdx.x * blockDim.x + threadIdx.x;
    size_t stride = (size_t)gridDim.x * blockDim.x;
    for (; i < n; i += stride) p[i] = 0.0f;
}

// ---------------- Edge MLP + scatter-add ----------------
// wave32 handles 16 edges; block = 8 waves = 128 edges.
__global__ __launch_bounds__(256) void edge_mlp_scatter(
    const float* __restrict__ x, const int* __restrict__ ei,
    const float* __restrict__ ea,
    const float* __restrict__ W1a, const float* __restrict__ b1a,
    const float* __restrict__ W1b, const float* __restrict__ b1b,
    float* __restrict__ agg, float* __restrict__ cnt, int E)
{
    __shared__ alignas(16) unsigned short sW1a[64 * 64];     // transposed [n][k]
    __shared__ alignas(16) unsigned short sW1b[64 * 64];     // transposed [n][k]
    __shared__ float sb1a[64], sb1b[64];
    __shared__ alignas(16) unsigned short sHid[8][16 * 64];  // per-wave [M][K] bf16
    __shared__ int sRow[8][16];

    int tid = threadIdx.x;
    for (int i = tid; i < 64 * 64; i += 256) {
        int k = i >> 6, n = i & 63;
        sW1a[n * 64 + k] = f2bf(W1a[i]);
        sW1b[n * 64 + k] = f2bf(W1b[i]);
    }
    if (tid < 64) { sb1a[tid] = b1a[tid]; sb1b[tid] = b1b[tid]; }
    __syncthreads();

    int wave = tid >> 5, lane = tid & 31;
    int m = lane & 15, half = lane >> 4;
    long long e0 = ((long long)blockIdx.x * 8 + wave) * 16;
    if (e0 >= E) return;
    long long e = e0 + m; if (e >= E) e = E - 1;
    int row = ei[e];
    int col = ei[(long long)E + e];
    if (lane < 16) sRow[wave][lane] = row;

    // A fragments: K 0..31 = x[col], K 32..63 = edge_attr[e]
    v16bf a0 = gather_row_bf16(x  + (size_t)col * F_NODE, half, 1.0f);
    v16bf a1 = gather_row_bf16(ea + (size_t)e   * F_EDGE, half, 1.0f);

    // GEMM1: [16,64] x [64,64], bias + relu, stage bf16 into LDS
    #pragma unroll
    for (int t = 0; t < 4; t++) {
        v8f c = {};
        const unsigned short* wb = &sW1a[(t * 16 + m) * 64];
        c = wmma_bf16(a0, ld8_8(wb + half * 8, wb + 16 + half * 8), c);
        c = wmma_bf16(a1, ld8_8(wb + 32 + half * 8, wb + 48 + half * 8), c);
        float bias = sb1a[t * 16 + m];
        #pragma unroll
        for (int r = 0; r < 8; r++) {
            float v = c[r] + bias; v = v > 0.0f ? v : 0.0f;
            sHid[wave][(r + half * 8) * 64 + t * 16 + m] = f2bf(v);
        }
    }

    // reload hidden (16x64) as two A fragments
    unsigned short* hb = sHid[wave];
    v16bf h0 = ld8_8(hb + m * 64 + half * 8,      hb + m * 64 + 16 + half * 8);
    v16bf h1 = ld8_8(hb + m * 64 + 32 + half * 8, hb + m * 64 + 48 + half * 8);

    // GEMM2: [16,64] x [64,64], bias, scatter atomic-add by destination node
    #pragma unroll
    for (int t = 0; t < 4; t++) {
        v8f c = {};
        const unsigned short* wb = &sW1b[(t * 16 + m) * 64];
        c = wmma_bf16(h0, ld8_8(wb + half * 8, wb + 16 + half * 8), c);
        c = wmma_bf16(h1, ld8_8(wb + 32 + half * 8, wb + 48 + half * 8), c);
        float bias = sb1b[t * 16 + m];
        int n = t * 16 + m;
        #pragma unroll
        for (int r = 0; r < 8; r++) {
            int M = r + half * 8;
            if (e0 + M < E)
                atomicAdd(&agg[(size_t)sRow[wave][M] * HID + n], c[r] + bias);
        }
    }
    if (lane < 16 && e0 + lane < E)
        atomicAdd(&cnt[sRow[wave][lane]], 1.0f);
}

// ---------------- Node MLP ----------------
__global__ __launch_bounds__(256) void node_mlp(
    const float* __restrict__ x,
    const float* __restrict__ agg, const float* __restrict__ cnt,
    const float* __restrict__ W2a, const float* __restrict__ b2a,
    const float* __restrict__ W2b, const float* __restrict__ b2b,
    float* __restrict__ out, int N)
{
    __shared__ alignas(16) unsigned short sW2a[64 * 96];   // transposed [n][k], K=96
    __shared__ alignas(16) unsigned short sW2b[32 * 64];   // transposed [n][k], K=64
    __shared__ float sb2a[64], sb2b[32];
    __shared__ alignas(16) unsigned short sHid[8][16 * 64];

    int tid = threadIdx.x;
    for (int i = tid; i < 96 * 64; i += 256) {
        int k = i >> 6, n = i & 63;                  // W2a is [96][64] row-major
        sW2a[n * 96 + k] = f2bf(W2a[i]);
    }
    for (int i = tid; i < 64 * 32; i += 256) {
        int k = i >> 5, n = i & 31;                  // W2b is [64][32] row-major
        sW2b[n * 64 + k] = f2bf(W2b[i]);
    }
    if (tid < 64) sb2a[tid] = b2a[tid];
    if (tid < 32) sb2b[tid] = b2b[tid];
    __syncthreads();

    int wave = tid >> 5, lane = tid & 31;
    int m = lane & 15, half = lane >> 4;
    long long n0 = ((long long)blockIdx.x * 8 + wave) * 16;
    if (n0 >= N) return;
    long long node = n0 + m; if (node >= N) node = N - 1;

    float c0 = cnt[node];
    float inv = 1.0f / (c0 > 1.0f ? c0 : 1.0f);      // scatter-mean normalization

    // h = [x (K 0..31), agg/cnt (K 32..95)]
    v16bf a0 = gather_row_bf16(x   + (size_t)node * F_NODE, half, 1.0f);
    const float* ar = agg + (size_t)node * HID;
    v16bf a1 = gather_row_bf16(ar,      half, inv);
    v16bf a2 = gather_row_bf16(ar + 32, half, inv);

    // GEMM1: [16,96] x [96,64], bias + relu
    #pragma unroll
    for (int t = 0; t < 4; t++) {
        v8f c = {};
        const unsigned short* wb = &sW2a[(t * 16 + m) * 96];
        c = wmma_bf16(a0, ld8_8(wb + half * 8, wb + 16 + half * 8), c);
        c = wmma_bf16(a1, ld8_8(wb + 32 + half * 8, wb + 48 + half * 8), c);
        c = wmma_bf16(a2, ld8_8(wb + 64 + half * 8, wb + 80 + half * 8), c);
        float bias = sb2a[t * 16 + m];
        #pragma unroll
        for (int r = 0; r < 8; r++) {
            float v = c[r] + bias; v = v > 0.0f ? v : 0.0f;
            sHid[wave][(r + half * 8) * 64 + t * 16 + m] = f2bf(v);
        }
    }

    unsigned short* hb = sHid[wave];
    v16bf h0 = ld8_8(hb + m * 64 + half * 8,      hb + m * 64 + 16 + half * 8);
    v16bf h1 = ld8_8(hb + m * 64 + 32 + half * 8, hb + m * 64 + 48 + half * 8);

    // GEMM2: [16,64] x [64,32], bias, store
    #pragma unroll
    for (int t = 0; t < 2; t++) {
        v8f c = {};
        const unsigned short* wb = &sW2b[(t * 16 + m) * 64];
        c = wmma_bf16(h0, ld8_8(wb + half * 8, wb + 16 + half * 8), c);
        c = wmma_bf16(h1, ld8_8(wb + 32 + half * 8, wb + 48 + half * 8), c);
        float bias = sb2b[t * 16 + m];
        int n = t * 16 + m;
        #pragma unroll
        for (int r = 0; r < 8; r++) {
            long long M = n0 + r + half * 8;
            if (M < N) out[(size_t)M * N_TGT + n] = c[r] + bias;
        }
    }
}

extern "C" void kernel_launch(void* const* d_in, const int* in_sizes, int n_in,
                              void* d_out, int out_size, void* d_ws, size_t ws_size,
                              hipStream_t stream)
{
    const float* x   = (const float*)d_in[0];
    const int*   ei  = (const int*)  d_in[1];   // edge_index flat [2,E]
    const float* ea  = (const float*)d_in[2];
    // d_in[3] = u, d_in[4] = batch : unused by the reference output path
    const float* W1a = (const float*)d_in[5];
    const float* b1a = (const float*)d_in[6];
    const float* W1b = (const float*)d_in[7];
    const float* b1b = (const float*)d_in[8];
    const float* W2a = (const float*)d_in[9];
    const float* b2a = (const float*)d_in[10];
    const float* W2b = (const float*)d_in[11];
    const float* b2b = (const float*)d_in[12];

    int N = in_sizes[0] / F_NODE;     // 100000
    int E = in_sizes[2] / F_EDGE;     // 1600000

    float* agg = (float*)d_ws;                    // [N, 64]
    float* cnt = agg + (size_t)N * HID;           // [N]
    size_t zn  = (size_t)N * HID + (size_t)N;

    zero_ws<<<1024, 256, 0, stream>>>(agg, zn);

    long long etiles = ((long long)E + 15) / 16;
    int eblocks = (int)((etiles + 7) / 8);
    edge_mlp_scatter<<<eblocks, 256, 0, stream>>>(x, ei, ea, W1a, b1a, W1b, b1b,
                                                  agg, cnt, E);

    long long ntiles = ((long long)N + 15) / 16;
    int nblocks = (int)((ntiles + 7) / 8);
    node_mlp<<<nblocks, 256, 0, stream>>>(x, agg, cnt, W2a, b2a, W2b, b2b,
                                          (float*)d_out, N);
}
